// MultiHeadSelfAttentionLayer_90967407329672
// MI455X (gfx1250) — matrix operand order repaired and checked by reading, not compiled, not run on hardware
//
#include <hip/hip_runtime.h>

// ---------------------------------------------------------------------------
// Multi-head self-attention, B=4 S=2048 D=1024 H=8 DH=128, scale = 1/H.
//  (1) weights fp32 -> bf16 transposed
//  (2) Q/K/V projections: 32x64 output per wave, 8 independent
//      v_wmma_f32_16x16x32_bf16 per k-step (x converted fp32->bf16 on the fly;
//      V stored transposed per head). Store branch hoisted (uniform).
//  (3) flash attention: 8 waves/block share one (b,h); K/V tiles DOUBLE-
//      BUFFERED in LDS via gfx1250 global_load_async_to_lds_b128 so staging
//      of tile j+1 overlaps compute of tile j. WMMA for QK^T and PV, online
//      softmax in fp32, P transposed C->A layout through per-wave LDS.
// ---------------------------------------------------------------------------

typedef unsigned short u16;
typedef unsigned int   u32;
typedef __bf16 bf16;
typedef bf16  v16bf __attribute__((ext_vector_type(16)));
typedef float v8f   __attribute__((ext_vector_type(8)));
typedef u32   u32x4 __attribute__((ext_vector_type(4)));
typedef float f32x4 __attribute__((ext_vector_type(4)));

#define N_B   4
#define N_S   2048
#define N_D   1024
#define N_H   8
#define N_DH  128
#define ATT_SCALE 0.125f   // 1 / N_HEADS
#define NEG_INF   (-3.0e38f)

union FragB { v16bf v; u16 u[16]; u32x4 q[2]; };

static __device__ inline u16 f32_to_bf16_bits(float f) {
    bf16 b = (bf16)f;
    return __builtin_bit_cast(u16, b);
}

// gfx1250 async copy: 16 bytes global -> LDS per lane, tracked by ASYNCcnt.
static __device__ inline void async_copy_b128(u32 lds_addr, const void* gaddr) {
    asm volatile("global_load_async_to_lds_b128 %0, %1, off"
                 :: "v"(lds_addr), "v"(gaddr) : "memory");
}

// ---------------------------------------------------------------------------
// Kernel 1: convert each weight (K x N fp32, row-major) to bf16 transposed
// (N x K), so WMMA B-fragments read two contiguous 16-byte runs of k per lane.
// grid: (4096, 3), block 256
// ---------------------------------------------------------------------------
__global__ __launch_bounds__(256) void wt_kernel(const float* __restrict__ Wq,
                                                 const float* __restrict__ Wk,
                                                 const float* __restrict__ Wv,
                                                 u16* __restrict__ WtAll) {
    const int idx = blockIdx.x * 256 + threadIdx.x;      // 0 .. 1M-1
    const int w   = blockIdx.y;                          // which weight
    const float* src = (w == 0) ? Wq : ((w == 1) ? Wk : Wv);
    const int k = idx >> 10;          // row of W
    const int n = idx & 1023;         // col of W
    const float v = src[idx];         // coalesced read  (= W[k][n])
    WtAll[(size_t)w * (N_D * N_D) + (size_t)n * N_D + k] = f32_to_bf16_bits(v);
}

// ---------------------------------------------------------------------------
// Kernel 2: Y = X @ W for Q/K/V. One wave computes a 32x64 strip: 2 A-frags
// x 4 B-frags -> 8 independent WMMAs per k-step.
// grid: (512, 3), block 256 (8 waves)
// ---------------------------------------------------------------------------
__global__ __launch_bounds__(256) void proj_kernel(const float* __restrict__ x,
                                                   const u16* __restrict__ WtAll,
                                                   u16* __restrict__ Qb,
                                                   u16* __restrict__ Kb,
                                                   u16* __restrict__ Vt) {
    const int lane  = threadIdx.x & 31;
    const int wid   = threadIdx.x >> 5;
    const int tile  = blockIdx.x * 8 + wid;   // 0 .. 4095
    const int mt    = tile >> 4;              // 256 row strips (32 rows)
    const int nt    = tile & 15;              // 16 col strips (64 cols)
    const int which = blockIdx.y;
    const int n16   = lane & 15;
    const int kh    = lane >> 4;

    const u16*   Wt    = WtAll + (size_t)which * (N_D * N_D);
    const float* arow0 = x + (size_t)(mt * 32 + n16) * N_D;
    const float* arow1 = x + (size_t)(mt * 32 + 16 + n16) * N_D;
    const u16*   brow[4];
    #pragma unroll
    for (int c = 0; c < 4; ++c)
        brow[c] = Wt + (size_t)(nt * 64 + c * 16 + n16) * N_D;

    v8f acc[2][4];
    const v8f vzero = {};
    #pragma unroll
    for (int s = 0; s < 2; ++s)
        #pragma unroll
        for (int c = 0; c < 4; ++c) acc[s][c] = vzero;

    for (int kk = 0; kk < 32; ++kk) {
        const int kb = kk * 32;
        FragB a[2];
        #pragma unroll
        for (int s = 0; s < 2; ++s) {
            const float* ar = (s == 0) ? arow0 : arow1;
            f32x4 a0 = *(const f32x4*)(ar + kb + 8 * kh);
            f32x4 a1 = *(const f32x4*)(ar + kb + 8 * kh + 4);
            f32x4 a2 = *(const f32x4*)(ar + kb + 16 + 8 * kh);
            f32x4 a3 = *(const f32x4*)(ar + kb + 16 + 8 * kh + 4);
            #pragma unroll
            for (int i = 0; i < 4; ++i) {
                a[s].u[i]      = f32_to_bf16_bits(a0[i]);
                a[s].u[4 + i]  = f32_to_bf16_bits(a1[i]);
                a[s].u[8 + i]  = f32_to_bf16_bits(a2[i]);
                a[s].u[12 + i] = f32_to_bf16_bits(a3[i]);
            }
        }
        FragB bf[4];
        #pragma unroll
        for (int c = 0; c < 4; ++c) {
            bf[c].q[0] = *(const u32x4*)(brow[c] + kb + 8 * kh);
            bf[c].q[1] = *(const u32x4*)(brow[c] + kb + 16 + 8 * kh);
        }
        #pragma unroll
        for (int s = 0; s < 2; ++s)
            #pragma unroll
            for (int c = 0; c < 4; ++c)
                acc[s][c] = __builtin_amdgcn_wmma_f32_16x16x32_bf16(
                    false, a[s].v, false, bf[c].v, (short)0, acc[s][c], false, false);
    }

    // ---- store: uniform branch hoisted outside the unrolled loops ---------
    const int row0 = mt * 32;
    const int col0 = nt * 64;
    if (which == 2) {
        #pragma unroll
        for (int s = 0; s < 2; ++s)
            #pragma unroll
            for (int c = 0; c < 4; ++c)
                #pragma unroll
                for (int r = 0; r < 8; ++r) {
                    const int row = row0 + s * 16 + r + 8 * kh;
                    const int col = col0 + c * 16 + n16;
                    const int bb = row >> 11, ss = row & 2047;
                    const int hh = col >> 7,  dh = col & 127;
                    Vt[(((size_t)(bb * N_H + hh)) * N_DH + dh) * N_S + ss] =
                        f32_to_bf16_bits(acc[s][c][r]);
                }
    } else {
        u16* dst = (which == 0) ? Qb : Kb;
        #pragma unroll
        for (int s = 0; s < 2; ++s)
            #pragma unroll
            for (int c = 0; c < 4; ++c)
                #pragma unroll
                for (int r = 0; r < 8; ++r) {
                    const int row = row0 + s * 16 + r + 8 * kh;
                    const int col = col0 + c * 16 + n16;
                    dst[(size_t)row * N_D + col] = f32_to_bf16_bits(acc[s][c][r]);
                }
    }
}

// ---------------------------------------------------------------------------
// Kernel 3: flash attention, double-buffered async K/V staging.
// grid: (16, 32), block 256 (8 waves)
// ---------------------------------------------------------------------------
__global__ __launch_bounds__(256) void attn_kernel(const u16* __restrict__ Qb,
                                                   const u16* __restrict__ Kb,
                                                   const u16* __restrict__ Vt,
                                                   float* __restrict__ out) {
    __shared__ __align__(16) u16 Kls[2][32 * 128];   // [buf][key][dh]   2x8 KB
    __shared__ __align__(16) u16 Vls[2][128 * 32];   // [buf][dh][key]   2x8 KB
    __shared__ __align__(16) u16 Pls[8][16 * 32];    // per-wave P tile    8 KB

    const int tid  = threadIdx.x;
    const int lane = tid & 31;
    const int wid  = tid >> 5;
    const int n16  = lane & 15;
    const int kh   = lane >> 4;
    const int bh   = blockIdx.y;           // 0..31
    const int b    = bh >> 3;
    const int h    = bh & 7;
    const int qt   = blockIdx.x * 8 + wid; // 0..127

    // ---- Q fragments for this wave's 16 rows (DH=128 -> 4 chunks of k=32)
    const u16* qrow = Qb + (size_t)(b * N_S + qt * 16 + n16) * N_D + h * N_DH;
    FragB qf[4];
    #pragma unroll
    for (int c = 0; c < 4; ++c) {
        const int kb = c * 32;
        qf[c].q[0] = *(const u32x4*)(qrow + kb + 8 * kh);
        qf[c].q[1] = *(const u32x4*)(qrow + kb + 16 + 8 * kh);
    }

    // ---- staging assignments (256 threads copy 8KB K + 8KB V per tile)
    const int kr = tid >> 3;               // key row 0..31
    const int kc = (tid & 7) * 16;         // element col of first 16B chunk
    const u16* kg_base = Kb + (size_t)(b * N_S + kr) * N_D + h * N_DH + kc;
    const u32  k_lds0  = (u32)(size_t)&Kls[0][kr * 128 + kc];
    const int vr = tid >> 1;               // dh row 0..127
    const int vc = (tid & 1) * 16;         // element col of first 16B chunk
    const u16* vg_base = Vt + (size_t)(bh * N_DH + vr) * N_S + vc;
    const u32  v_lds0  = (u32)(size_t)&Vls[0][vr * 32 + vc];
    const u32  KBUF_STRIDE = 32 * 128 * 2; // bytes between double buffers
    const u32  VBUF_STRIDE = 128 * 32 * 2;

    float m_r[8], l_r[8];
    v8f   oacc[8];
    const v8f vzero = {};
    #pragma unroll
    for (int r = 0; r < 8; ++r) { m_r[r] = NEG_INF; l_r[r] = 0.0f; oacc[r] = vzero; }

    u16* myP = Pls[wid];

    // prologue: stage tile 0 into buffer 0
    async_copy_b128(k_lds0,      kg_base);
    async_copy_b128(k_lds0 + 16, kg_base + 8);
    async_copy_b128(v_lds0,      vg_base);
    async_copy_b128(v_lds0 + 16, vg_base + 8);

    for (int j = 0; j < 64; ++j) {         // 64 key tiles of 32 keys
        // wait for this wave's outstanding staging, then block-wide barrier
        asm volatile("s_wait_asynccnt 0" ::: "memory");
        __syncthreads();

        // stage tile j+1 into the other buffer (overlaps compute below)
        if (j < 63) {
            const int jn = j + 1;
            const u32 ko = k_lds0 + (jn & 1) * KBUF_STRIDE;
            const u32 vo = v_lds0 + (jn & 1) * VBUF_STRIDE;
            async_copy_b128(ko,      kg_base + (size_t)jn * 32 * N_D);
            async_copy_b128(ko + 16, kg_base + (size_t)jn * 32 * N_D + 8);
            async_copy_b128(vo,      vg_base + jn * 32);
            async_copy_b128(vo + 16, vg_base + jn * 32 + 8);
        }

        const u16* Kbuf = Kls[j & 1];
        const u16* Vbuf = Vls[j & 1];

        // ---- scores: S[16x32] = Q_tile @ K_tile^T -------------------------
        v8f s0 = vzero, s1 = vzero;
        const u16* kls0 = Kbuf + n16 * 128;
        const u16* kls1 = Kbuf + (16 + n16) * 128;
        #pragma unroll
        for (int c = 0; c < 4; ++c) {
            const int kb = c * 32;
            FragB b0, b1;
            b0.q[0] = *(const u32x4*)(kls0 + kb + 8 * kh);
            b0.q[1] = *(const u32x4*)(kls0 + kb + 16 + 8 * kh);
            b1.q[0] = *(const u32x4*)(kls1 + kb + 8 * kh);
            b1.q[1] = *(const u32x4*)(kls1 + kb + 16 + 8 * kh);
            s0 = __builtin_amdgcn_wmma_f32_16x16x32_bf16(false, qf[c].v, false, b0.v,
                                                         (short)0, s0, false, false);
            s1 = __builtin_amdgcn_wmma_f32_16x16x32_bf16(false, qf[c].v, false, b1.v,
                                                         (short)0, s1, false, false);
        }

        // ---- online softmax over 32 new columns ---------------------------
        float corr[8];
        #pragma unroll
        for (int r = 0; r < 8; ++r) {
            const float v0 = s0[r] * ATT_SCALE;
            const float v1 = s1[r] * ATT_SCALE;
            float tm = fmaxf(v0, v1);
            tm = fmaxf(tm, __shfl_xor(tm, 1, 32));
            tm = fmaxf(tm, __shfl_xor(tm, 2, 32));
            tm = fmaxf(tm, __shfl_xor(tm, 4, 32));
            tm = fmaxf(tm, __shfl_xor(tm, 8, 32));   // row max within 16 lanes
            const float mnew = fmaxf(m_r[r], tm);
            corr[r] = __expf(m_r[r] - mnew);
            m_r[r]  = mnew;
            const float p0 = __expf(v0 - mnew);
            const float p1 = __expf(v1 - mnew);
            float ts = p0 + p1;
            ts += __shfl_xor(ts, 1, 32);
            ts += __shfl_xor(ts, 2, 32);
            ts += __shfl_xor(ts, 4, 32);
            ts += __shfl_xor(ts, 8, 32);
            l_r[r] = l_r[r] * corr[r] + ts;
            // stage P in LDS (C layout -> row-major 16x32 bf16)
            const int m = r + 8 * kh;
            myP[m * 32 + n16]      = f32_to_bf16_bits(p0);
            myP[m * 32 + 16 + n16] = f32_to_bf16_bits(p1);
        }

        // rescale running output accumulator
        #pragma unroll
        for (int ct = 0; ct < 8; ++ct) {
            #pragma unroll
            for (int r = 0; r < 8; ++r) oacc[ct][r] *= corr[r];
        }

        asm volatile("s_wait_dscnt 0" ::: "memory");

        // reload P as A-matrix fragment (16x32 bf16)
        FragB pf;
        {
            const u16* prow = myP + n16 * 32;
            pf.q[0] = *(const u32x4*)(prow + 8 * kh);
            pf.q[1] = *(const u32x4*)(prow + 16 + 8 * kh);
        }

        // ---- PV: O[16x128] += P[16x32] @ V[32x128] ------------------------
        #pragma unroll
        for (int ct = 0; ct < 8; ++ct) {
            const u16* vrow = Vbuf + (ct * 16 + n16) * 32;
            FragB vf;
            vf.q[0] = *(const u32x4*)(vrow + 8 * kh);
            vf.q[1] = *(const u32x4*)(vrow + 16 + 8 * kh);
            oacc[ct] = __builtin_amdgcn_wmma_f32_16x16x32_bf16(false, pf.v, false, vf.v,
                                                               (short)0, oacc[ct], false, false);
        }
    }

    // ---- epilogue: divide by row sums, store fp32 -------------------------
    #pragma unroll
    for (int ct = 0; ct < 8; ++ct) {
        #pragma unroll
        for (int r = 0; r < 8; ++r) {
            const int m   = r + 8 * kh;
            const int row = b * N_S + qt * 16 + m;
            const int col = h * N_DH + ct * 16 + n16;
            out[(size_t)row * N_D + col] = oacc[ct][r] / l_r[r];
        }
    }
}

// ---------------------------------------------------------------------------
extern "C" void kernel_launch(void* const* d_in, const int* in_sizes, int n_in,
                              void* d_out, int out_size, void* d_ws, size_t ws_size,
                              hipStream_t stream) {
    (void)in_sizes; (void)n_in; (void)out_size; (void)ws_size;
    const float* x  = (const float*)d_in[0];
    const float* Wq = (const float*)d_in[1];
    const float* Wk = (const float*)d_in[2];
    const float* Wv = (const float*)d_in[3];
    float* out = (float*)d_out;

    // workspace layout (bf16): Wt[3*1M] | Q[8M] | K[8M] | Vt[8M]  = ~56.6 MB
    u16* Wt = (u16*)d_ws;
    u16* Qb = Wt + (size_t)3 * N_D * N_D;
    u16* Kb = Qb + (size_t)(N_B * N_S) * N_D;
    u16* Vt = Kb + (size_t)(N_B * N_S) * N_D;

    wt_kernel  <<<dim3(4096, 3), 256, 0, stream>>>(Wq, Wk, Wv, Wt);
    proj_kernel<<<dim3(512, 3),  256, 0, stream>>>(x, Wt, Qb, Kb, Vt);
    attn_kernel<<<dim3(16, 32),  256, 0, stream>>>(Qb, Kb, Vt, out);
}
